// BERT_936302870555
// MI455X (gfx1250) — matrix operand order repaired and checked by reading, not compiled, not run on hardware
//
#include <hip/hip_runtime.h>
#include <hip/hip_bf16.h>
#include <math.h>

// ---------------------------------------------------------------------------
// Fused 4-branch BERT stack for MI455X (gfx1250, wave32, WMMA).
// One workgroup = one (branch, batch); everything resident in <=320KB LDS.
// All GEMMs via v_wmma_f32_16x16x32_bf16; fragment feeds are unconditional
// 16B LDS vector loads (K-contiguous layouts), no per-element branching.
// ---------------------------------------------------------------------------

typedef __attribute__((ext_vector_type(16))) __bf16 v16bf;
typedef __attribute__((ext_vector_type(8)))  __bf16 v8bf;
typedef __attribute__((ext_vector_type(8)))  float  v8f;

#define DMODEL 36
#define SEQ    600
#define LPAD   608
#define MTILES 38
#define ZSTR   37          // f32 residual row stride
#define BSTR   40          // bf16 activation row stride (80B = 16B-aligned rows)
#define DFF    144
#define VSTR   624         // vT: [40 dims][624 keys]
#define WSTR   64          // transposed weight K-stride (Wq/Wk/Wv/Wo/W1)
#define W2STR  160         // transposed W2 K-stride

// LDS byte offsets (total 321,088 B <= 320KB)
#define OFF_Z   0          // f32 [608][37]                       89,984 B
#define OFF_Y   89984      // bf16 [608][40] + 64 slack           48,768 B
#define OFF_Q   138752     // bf16 [608][40] + 96 slack           48,832 B
#define OFF_K   187584     // bf16 [608][40] + 64 slack           48,768 B
#define OFF_V   236352     // bf16 vT [40][624]                   49,920 B
#define OFF_W   286272     // bf16 transposed weights             33,792 B
#define OFF_CT  320064     // f32 SE-gate scratch                  1,024 B
#define SHMEM_BYTES 321088
#define PBUF_ELEM_OFF 9216 // pbuf overlays wls spare during attention

__device__ __constant__ int kHeads[24] = {1,2,3,4,6,9,12,18,36,
                                          1,2,3,4,6,
                                          6,9,12,18,36,
                                          3,4,6,9,12};
__device__ __constant__ int kStart[4] = {0,9,14,19};
__device__ __constant__ int kEnd[4]   = {9,14,19,24};

__device__ inline v8f v8f_zero() {
  v8f z;
#pragma unroll
  for (int i = 0; i < 8; ++i) z[i] = 0.0f;
  return z;
}

__device__ inline v16bf cat8(v8bf lo, v8bf hi8) {
  return __builtin_shufflevector(lo, hi8, 0,1,2,3,4,5,6,7,8,9,10,11,12,13,14,15);
}

// ---- Wide A fragment (16x32 bf16). Requires stride%8==0 && col0%8==0.
// Lane layout: m=lane&15; elems = row[col0 + 16*hi .. +15]
__device__ inline v16bf load_a_w(const __bf16* base, int stride, int row0, int col0) {
  const int lane = threadIdx.x & 31;
  const int m = lane & 15, hi = lane >> 4;
  const __bf16* rp = base + (row0 + m) * stride + col0 + hi * 16;
  v8bf lo = *(const v8bf*)(rp);
  v8bf hh = *(const v8bf*)(rp + 8);
  return cat8(lo, hh);
}

// ---- Elementwise A fragment (arbitrary col0; garbage K killed by zeroed B).
__device__ inline v16bf load_a_e(const __bf16* base, int stride, int row0, int col0) {
  const int lane = threadIdx.x & 31;
  const int m = lane & 15, hi = lane >> 4;
  const __bf16* rp = base + (row0 + m) * stride + col0;
  v16bf a;
#pragma unroll
  for (int v = 0; v < 8; ++v) {
    int k0 = 2 * v + (hi ? 16 : 0);
    a[2 * v]     = rp[k0];
    a[2 * v + 1] = rp[k0 + 1];
  }
  return a;
}

// ---- Wide B fragment from K-contiguous (transposed) source:
// element (k,n) = baseT[(col0+n)*stride + k0 + k]. stride%8==0 && k0%8==0.
__device__ inline v16bf load_b_w(const __bf16* baseT, int stride, int k0, int col0) {
  const int lane = threadIdx.x & 31;
  const int n = lane & 15, hi = lane >> 4;
  const __bf16* rp = baseT + (col0 + n) * stride + k0 + hi * 16;
  v8bf lo = *(const v8bf*)(rp);
  v8bf hh = *(const v8bf*)(rp + 8);
  return cat8(lo, hh);
}

// ---- Wide B fragment from vT with dim-row clamp (clamp keeps alignment).
__device__ inline v16bf load_bv_w(const __bf16* vT, int krow0, int col0) {
  const int lane = threadIdx.x & 31;
  const int n = lane & 15, hi = lane >> 4;
  int nr = col0 + n; nr = (nr > 39) ? 39 : nr;   // one v_min, garbage never stored
  const __bf16* rp = vT + nr * VSTR + krow0 + hi * 16;
  v8bf lo = *(const v8bf*)(rp);
  v8bf hh = *(const v8bf*)(rp + 8);
  return cat8(lo, hh);
}

// ---- B^T fragment for Q·K^T (K-contiguous per lane) with select-zero on K.
// klen is compile-time after unrolling (templated DK) -> selects mostly fold.
__device__ inline v16bf load_bt_e(const __bf16* base, int stride, int nrow0,
                                  int col0, int klen) {
  const int lane = threadIdx.x & 31;
  const int n = lane & 15, hi = lane >> 4;
  const __bf16* rp = base + (nrow0 + n) * stride + col0 + hi * 16;
  const int kbase = hi * 16;
  v16bf b;
#pragma unroll
  for (int i = 0; i < 16; ++i) {
    __bf16 t = rp[i];                       // unconditional (slack-covered)
    b[i] = (kbase + i < klen) ? t : (__bf16)0.0f;
  }
  return b;
}

__device__ inline v8f wmma_bf16(v16bf a, v16bf b, v8f c) {
  return __builtin_amdgcn_wmma_f32_16x16x32_bf16(
      false, a, false, b, (short)0, c, false, false);
}

__device__ inline void layer_norm_rows(const float* z, __bf16* y,
                                       const float* g, const float* bta, int tid) {
  for (int l = tid; l < LPAD; l += 256) {
    const float* zr = z + l * ZSTR;
    float m = 0.0f;
#pragma unroll
    for (int c = 0; c < DMODEL; ++c) m += zr[c];
    m *= (1.0f / DMODEL);
    float var = 0.0f;
#pragma unroll
    for (int c = 0; c < DMODEL; ++c) { float d = zr[c] - m; var += d * d; }
    var *= (1.0f / DMODEL);
    float inv = rsqrtf(var + 1e-6f);
    __bf16* yr = y + l * BSTR;
#pragma unroll
    for (int c = 0; c < DMODEL; ++c)
      yr[c] = (__bf16)(g[c] * (zr[c] - m) * inv + bta[c]);
  }
}

// ---- Flash attention for one layer, specialized on head dim DK ------------
template <int DK>
__device__ void attention_layer(const __bf16* qb, const __bf16* kb,
                                const __bf16* vT, __bf16* ybuf, __bf16* pw,
                                int h, int wave, int nlo, int hi) {
  constexpr int NTO = (DK + 15) / 16;     // O column tiles
  constexpr int NKC = (DK + 31) / 32;     // K chunks for S
  const float scale = rsqrtf((float)DK);
  for (int t = wave; t < h * MTILES; t += 8) {
    const int head = t / MTILES, qt = t % MTILES;
    const int cbase = head * DK;
    float mrun[8], lrun[8];
    v8f oacc[NTO];
#pragma unroll
    for (int r = 0; r < 8; ++r) { mrun[r] = -INFINITY; lrun[r] = 0.0f; }
#pragma unroll
    for (int j = 0; j < NTO; ++j) oacc[j] = v8f_zero();

    for (int kt = 0; kt < MTILES; ++kt) {
      // S = (Q K^T) * scale
      v8f sacc = v8f_zero();
#pragma unroll
      for (int c = 0; c < NKC; ++c) {
        const int k0 = 32 * c;
        const int klen = (DK - k0 > 32) ? 32 : (DK - k0);
        v16bf a  = load_a_e(qb, BSTR, qt * 16, cbase + k0);
        v16bf bt = load_bt_e(kb, BSTR, kt * 16, cbase + k0, klen);
        sacc = wmma_bf16(a, bt, sacc);
      }
      // online softmax; row = r+8*hi, key cols on lanes of each half-wave
      const int key = kt * 16 + nlo;
      const bool kvalid = (key < SEQ);
      float pv[8];
#pragma unroll
      for (int r = 0; r < 8; ++r) {
        float sv = kvalid ? sacc[r] * scale : -3.0e38f;
        float mx = sv;
        mx = fmaxf(mx, __shfl_xor(mx, 1));
        mx = fmaxf(mx, __shfl_xor(mx, 2));
        mx = fmaxf(mx, __shfl_xor(mx, 4));
        mx = fmaxf(mx, __shfl_xor(mx, 8));
        float nm = fmaxf(mrun[r], mx);
        float p  = __expf(sv - nm);
        float rs = p;
        rs += __shfl_xor(rs, 1);
        rs += __shfl_xor(rs, 2);
        rs += __shfl_xor(rs, 4);
        rs += __shfl_xor(rs, 8);
        float alpha = __expf(mrun[r] - nm);
        lrun[r] = lrun[r] * alpha + rs;
        mrun[r] = nm;
        pv[r] = p;
#pragma unroll
        for (int j = 0; j < NTO; ++j) oacc[j][r] *= alpha;
      }
      // P tile -> wave-private LDS (cols 16..31 are zero; kill V row garbage)
#pragma unroll
      for (int r = 0; r < 8; ++r)
        pw[(r + 8 * hi) * 32 + nlo] = (__bf16)pv[r];
      // O += P @ V
      v16bf pa = load_a_w(pw, 32, 0, 0);
#pragma unroll
      for (int j = 0; j < NTO; ++j) {
        v16bf vf = load_bv_w(vT, kt * 16, cbase + j * 16);
        oacc[j] = wmma_bf16(pa, vf, oacc[j]);
      }
    }
    // normalize + write O into ybuf (guard only the ragged last tile)
#pragma unroll
    for (int j = 0; j < NTO; ++j) {
      const int col = cbase + j * 16 + nlo;
      const bool ok = (j < NTO - 1) || (nlo < DK - (NTO - 1) * 16);
      if (ok) {
#pragma unroll
        for (int r = 0; r < 8; ++r)
          ybuf[(qt * 16 + r + 8 * hi) * BSTR + col] =
              (__bf16)(oacc[j][r] / lrun[r]);
      }
    }
  }
}

__global__ void __launch_bounds__(256)
bert_stack_kernel(const float* __restrict__ x,
                  const float* __restrict__ ln1_g, const float* __restrict__ ln1_b,
                  const float* __restrict__ Wq, const float* __restrict__ bq,
                  const float* __restrict__ Wk, const float* __restrict__ bk,
                  const float* __restrict__ Wv, const float* __restrict__ bv,
                  const float* __restrict__ Wo, const float* __restrict__ bo,
                  const float* __restrict__ ln2_g, const float* __restrict__ ln2_b,
                  const float* __restrict__ W1, const float* __restrict__ b1,
                  const float* __restrict__ W2, const float* __restrict__ b2,
                  float* __restrict__ ws) {
  extern __shared__ char smem[];
  float*  z    = (float*)(smem + OFF_Z);
  __bf16* ybuf = (__bf16*)(smem + OFF_Y);
  __bf16* qb   = (__bf16*)(smem + OFF_Q);
  __bf16* kb   = (__bf16*)(smem + OFF_K);
  __bf16* vT   = (__bf16*)(smem + OFF_V);
  __bf16* wls  = (__bf16*)(smem + OFF_W);
  float*  ctl  = (float*)(smem + OFF_CT);
  __bf16* pbuf = wls + PBUF_ELEM_OFF;       // 8 waves * [16][32]

  const int s    = blockIdx.x;
  const int b    = blockIdx.y;
  const int tid  = threadIdx.x;
  const int wave = tid >> 5;
  const int lane = tid & 31;
  const int nlo  = lane & 15;
  const int hi   = lane >> 4;

  // ---- Phase 0: h0 = x^T + sinusoidal PE (pad rows/cols zero) ----
  for (int i = tid; i < LPAD * ZSTR; i += 256) {
    int l = i / ZSTR, c = i % ZSTR;
    float v = 0.0f;
    if (l < SEQ && c < DMODEL) {
      float fi  = (float)(c & ~1);
      float ang = (float)l * __expf(-0.2558427881f * fi);  // ln(1e4)/36
      float pe  = (c & 1) ? __cosf(ang) : __sinf(ang);
      v = x[(b * DMODEL + c) * SEQ + l] + pe;
    }
    z[i] = v;
  }
  __syncthreads();

  for (int blk = kStart[s]; blk < kEnd[s]; ++blk) {
    const int h  = kHeads[blk];
    const int dk = DMODEL / h;
    const float* wqp = Wq + blk * DMODEL * DMODEL;
    const float* wkp = Wk + blk * DMODEL * DMODEL;
    const float* wvp = Wv + blk * DMODEL * DMODEL;
    const float* wop = Wo + blk * DMODEL * DMODEL;

    if (blk + 1 < kEnd[s]) {
      __builtin_prefetch(Wq + (blk + 1) * DMODEL * DMODEL, 0, 1);
      __builtin_prefetch(W1 + (blk + 1) * DMODEL * DFF, 0, 1);
    }

    // ---- stage Wq/Wk/Wv transposed [48 n][64 k] (zero-padded) + LN1 ----
    for (int i = tid; i < 3 * 48 * WSTR; i += 256) {
      int mat = i / (48 * WSTR), r = i % (48 * WSTR);
      int n = r >> 6, k = r & 63;
      const float* src = (mat == 0) ? wqp : (mat == 1) ? wkp : wvp;
      wls[i] = (n < DMODEL && k < DMODEL) ? (__bf16)src[k * DMODEL + n]
                                          : (__bf16)0.0f;
    }
    for (int i = tid; i < 8 * 512; i += 256) pbuf[i] = (__bf16)0.0f;
    layer_norm_rows(z, ybuf, ln1_g + blk * DMODEL, ln1_b + blk * DMODEL, tid);
    __syncthreads();

    // ---- QKV GEMMs (K=36 padded; weight zeros kill activation garbage) ----
    for (int t = wave; t < 3 * MTILES * 3; t += 8) {
      int mat = t / (MTILES * 3);
      int rem = t % (MTILES * 3);
      int mt = rem / 3, nt = rem % 3;
      const __bf16* wT = wls + mat * 48 * WSTR;
      v8f acc = v8f_zero();
#pragma unroll
      for (int c = 0; c < 2; ++c) {
        v16bf a  = load_a_w(ybuf, BSTR, mt * 16, 32 * c);
        v16bf bb = load_b_w(wT, WSTR, 32 * c, nt * 16);
        acc = wmma_bf16(a, bb, acc);
      }
      const float* bias = (mat == 0) ? bq : (mat == 1) ? bk : bv;
      int col = nt * 16 + nlo;
      if (col < DMODEL) {
        float bsv = bias[blk * DMODEL + col];
        if (mat == 2) {   // V stored transposed: vT[dim][key]
#pragma unroll
          for (int r = 0; r < 8; ++r)
            vT[col * VSTR + (mt * 16 + r + 8 * hi)] = (__bf16)(acc[r] + bsv);
        } else {
          __bf16* dst = (mat == 0) ? qb : kb;
#pragma unroll
          for (int r = 0; r < 8; ++r)
            dst[(mt * 16 + r + 8 * hi) * BSTR + col] = (__bf16)(acc[r] + bsv);
        }
      }
    }
    __syncthreads();

    // ---- Flash attention, specialized on head dim ----
    switch (dk) {
      case 36: attention_layer<36>(qb, kb, vT, ybuf, pbuf + wave * 512, h, wave, nlo, hi); break;
      case 18: attention_layer<18>(qb, kb, vT, ybuf, pbuf + wave * 512, h, wave, nlo, hi); break;
      case 12: attention_layer<12>(qb, kb, vT, ybuf, pbuf + wave * 512, h, wave, nlo, hi); break;
      case  9: attention_layer< 9>(qb, kb, vT, ybuf, pbuf + wave * 512, h, wave, nlo, hi); break;
      case  6: attention_layer< 6>(qb, kb, vT, ybuf, pbuf + wave * 512, h, wave, nlo, hi); break;
      case  4: attention_layer< 4>(qb, kb, vT, ybuf, pbuf + wave * 512, h, wave, nlo, hi); break;
      case  3: attention_layer< 3>(qb, kb, vT, ybuf, pbuf + wave * 512, h, wave, nlo, hi); break;
      default: attention_layer< 1>(qb, kb, vT, ybuf, pbuf + wave * 512, h, wave, nlo, hi); break;
    }
    __syncthreads();

    // ---- stage Wo transposed ----
    for (int i = tid; i < 48 * WSTR; i += 256) {
      int n = i >> 6, k = i & 63;
      wls[i] = (n < DMODEL && k < DMODEL) ? (__bf16)wop[k * DMODEL + n]
                                          : (__bf16)0.0f;
    }
    __syncthreads();

    // ---- O projection + residual ----
    for (int t = wave; t < MTILES * 3; t += 8) {
      int mt = t / 3, nt = t % 3;
      v8f acc = v8f_zero();
#pragma unroll
      for (int c = 0; c < 2; ++c) {
        v16bf a  = load_a_w(ybuf, BSTR, mt * 16, 32 * c);
        v16bf bb = load_b_w(wls, WSTR, 32 * c, nt * 16);
        acc = wmma_bf16(a, bb, acc);
      }
      int col = nt * 16 + nlo;
      if (col < DMODEL) {
        float bov = bo[blk * DMODEL + col];
#pragma unroll
        for (int r = 0; r < 8; ++r)
          z[(mt * 16 + r + 8 * hi) * ZSTR + col] += acc[r] + bov;
      }
    }
    __syncthreads();

    // ---- LN2 + stage W1^T [144][64], W2^T [48][160] ----
    for (int i = tid; i < DFF * WSTR; i += 256) {
      int n = i >> 6, k = i & 63;
      wls[i] = (k < DMODEL) ? (__bf16)W1[blk * DMODEL * DFF + k * DFF + n]
                            : (__bf16)0.0f;
    }
    for (int i = tid; i < 48 * W2STR; i += 256) {
      int n = i / W2STR, k = i % W2STR;
      wls[DFF * WSTR + i] =
          (n < DMODEL && k < DFF)
              ? (__bf16)W2[blk * DFF * DMODEL + k * DMODEL + n]
              : (__bf16)0.0f;
    }
    layer_norm_rows(z, ybuf, ln2_g + blk * DMODEL, ln2_b + blk * DMODEL, tid);
    __syncthreads();

    // ---- FFN: per-wave 16-row tiles, hidden in wave-private LDS ----
    {
      const __bf16* w1T = wls;
      const __bf16* w2T = wls + DFF * WSTR;
      __bf16* hid = qb + wave * 16 * 160;    // q is dead here
      for (int mt = wave; mt < MTILES; mt += 8) {
#pragma unroll
        for (int nt = 0; nt < 9; ++nt) {     // 144 hidden cols
          v8f acc = v8f_zero();
#pragma unroll
          for (int c = 0; c < 2; ++c) {
            v16bf a  = load_a_w(ybuf, BSTR, mt * 16, 32 * c);
            v16bf bb = load_b_w(w1T, WSTR, 32 * c, nt * 16);
            acc = wmma_bf16(a, bb, acc);
          }
          int col = nt * 16 + nlo;
          float b1v = b1[blk * DFF + col];
#pragma unroll
          for (int r = 0; r < 8; ++r) {
            float v = acc[r] + b1v;
            float g = 0.5f * v *
                      (1.0f + tanhf(0.7978845608f * (v + 0.044715f * v * v * v)));
            hid[(r + 8 * hi) * 160 + col] = (__bf16)g;
          }
        }
#pragma unroll
        for (int nt = 0; nt < 3; ++nt) {
          v8f acc = v8f_zero();
#pragma unroll
          for (int c = 0; c < 5; ++c) {      // K=144 padded to 160 (W2 zeros)
            v16bf a  = load_a_w(hid, 160, 0, 32 * c);
            v16bf bb = load_b_w(w2T, W2STR, 32 * c, nt * 16);
            acc = wmma_bf16(a, bb, acc);
          }
          int col = nt * 16 + nlo;
          if (col < DMODEL) {
            float b2v = b2[blk * DMODEL + col];
#pragma unroll
            for (int r = 0; r < 8; ++r)
              z[(mt * 16 + r + 8 * hi) * ZSTR + col] += acc[r] + b2v;
          }
        }
      }
    }
    __syncthreads();
  }

  // ---- SE gate: sigmoid(mean over L) * z -> gated branch to workspace ----
  float* gpart = ctl;          // [36][4]
  float* gate  = ctl + 160;    // [36]
  if (tid < 144) {
    int c = tid >> 2, p = tid & 3;
    float sum = 0.0f;
    for (int l = p; l < SEQ; l += 4) sum += z[l * ZSTR + c];
    gpart[c * 4 + p] = sum;
  }
  __syncthreads();
  if (tid < DMODEL) {
    float m = (gpart[tid * 4] + gpart[tid * 4 + 1] +
               gpart[tid * 4 + 2] + gpart[tid * 4 + 3]) * (1.0f / SEQ);
    gate[tid] = 1.0f / (1.0f + __expf(-m));
  }
  __syncthreads();
  for (int i = tid; i < DMODEL * SEQ; i += 256) {
    int c = i / SEQ, l = i % SEQ;
    ws[((b * 4 + s) * DMODEL + c) * SEQ + l] = gate[c] * z[l * ZSTR + c];
  }
}

// ---- Conv1d(144->36, k=3, pad=1) + BatchNorm(eval) + ReLU ----
__global__ void __launch_bounds__(256)
conv_bn_relu_kernel(const float* __restrict__ ws,
                    const float* __restrict__ conv_w,
                    const float* __restrict__ conv_b,
                    const float* __restrict__ bn_g, const float* __restrict__ bn_b,
                    const float* __restrict__ bn_mean, const float* __restrict__ bn_var,
                    float* __restrict__ out) {
  int idx = blockIdx.x * blockDim.x + threadIdx.x;
  if (idx >= 4 * DMODEL * SEQ) return;
  int l  = idx % SEQ;
  int t  = idx / SEQ;
  int oc = t % DMODEL;
  int bb = t / DMODEL;
  float sum = conv_b[oc];
  for (int ic = 0; ic < 4 * DMODEL; ++ic) {
    const float* xr = ws + (bb * 4 * DMODEL + ic) * SEQ;
    const float* wr = conv_w + (oc * 4 * DMODEL + ic) * 3;
    if (l > 0)       sum += wr[0] * xr[l - 1];
                     sum += wr[1] * xr[l];
    if (l < SEQ - 1) sum += wr[2] * xr[l + 1];
  }
  float y = bn_g[oc] * (sum - bn_mean[oc]) * rsqrtf(bn_var[oc] + 1e-5f) + bn_b[oc];
  out[idx] = fmaxf(y, 0.0f);
}

extern "C" void kernel_launch(void* const* d_in, const int* in_sizes, int n_in,
                              void* d_out, int out_size, void* d_ws, size_t ws_size,
                              hipStream_t stream) {
  const float* x     = (const float*)d_in[0];
  const float* ln1_g = (const float*)d_in[1];
  const float* ln1_b = (const float*)d_in[2];
  const float* Wq    = (const float*)d_in[3];
  const float* bq    = (const float*)d_in[4];
  const float* Wk    = (const float*)d_in[5];
  const float* bk    = (const float*)d_in[6];
  const float* Wv    = (const float*)d_in[7];
  const float* bv    = (const float*)d_in[8];
  const float* Wo    = (const float*)d_in[9];
  const float* bo    = (const float*)d_in[10];
  const float* ln2_g = (const float*)d_in[11];
  const float* ln2_b = (const float*)d_in[12];
  const float* W1    = (const float*)d_in[13];
  const float* b1    = (const float*)d_in[14];
  const float* W2    = (const float*)d_in[15];
  const float* b2    = (const float*)d_in[16];
  const float* conv_w = (const float*)d_in[17];
  const float* conv_b = (const float*)d_in[18];
  const float* bn_g  = (const float*)d_in[19];
  const float* bn_b  = (const float*)d_in[20];
  const float* bn_m  = (const float*)d_in[21];
  const float* bn_v  = (const float*)d_in[22];
  float* ws  = (float*)d_ws;
  float* out = (float*)d_out;

  (void)hipFuncSetAttribute((const void*)bert_stack_kernel,
                            hipFuncAttributeMaxDynamicSharedMemorySize,
                            SHMEM_BYTES);

  dim3 grid1(4, 4);   // (stack, batch)
  bert_stack_kernel<<<grid1, 256, SHMEM_BYTES, stream>>>(
      x, ln1_g, ln1_b, Wq, bq, Wk, bk, Wv, bv, Wo, bo,
      ln2_g, ln2_b, W1, b1, W2, b2, ws);

  int nout = 4 * DMODEL * SEQ;
  conv_bn_relu_kernel<<<(nout + 255) / 256, 256, 0, stream>>>(
      ws, conv_w, conv_b, bn_g, bn_b, bn_m, bn_v, out);
}
